// MemAttention_52312701665913
// MI455X (gfx1250) — matrix operand order repaired and checked by reading, not compiled, not run on hardware
//
#include <hip/hip_runtime.h>
#include <hip/hip_fp16.h>

typedef _Float16 h16;
typedef _Float16 v16h __attribute__((ext_vector_type(16)));
typedef float    v8f  __attribute__((ext_vector_type(8)));

#define GF_RELU 1
#define GF_RES  2
#define GF_OUTF 4
#define GF_OUTH 8
#define GF_OUTT 16

__device__ inline v8f wmma_f16(v16h a, v16h b, v8f c) {
  return __builtin_amdgcn_wmma_f32_16x16x32_f16(false, a, false, b, (short)0, c, false, false);
}

// Async DMA: one 16-byte chunk per lane, global -> LDS (ASYNCcnt-tracked)
__device__ inline void async_cp16(const h16* g, unsigned lds_addr) {
  asm volatile("global_load_async_to_lds_b128 %0, %1, off"
               :: "v"(lds_addr), "v"((unsigned long long)(uintptr_t)g)
               : "memory");
}
__device__ inline unsigned lds_lo32(const void* p) {
  return (unsigned)(uintptr_t)p;   // flat->LDS: hardware truncates to addr[31:0]
}

// ---------------------------------------------------------------------------
// Weight convert: W[K][N] f32 -> Wt[N][K] f16 (K-major for contiguous B-frags)
// ---------------------------------------------------------------------------
__global__ void wt_convert(const float* __restrict__ W, h16* __restrict__ Wt, int K, int N) {
  int idx = blockIdx.x * 256 + threadIdx.x;
  if (idx >= K * N) return;
  int k = idx / N, n = idx % N;
  Wt[(size_t)n * K + k] = (h16)W[idx];
}

// ---------------------------------------------------------------------------
// x[b][s][c] = feat[c][s] - no_mem[c] + 0.1*pos[s][c]      (f32 residual path)
// ---------------------------------------------------------------------------
__global__ void prep_x(const float* __restrict__ feat, const float* __restrict__ pos,
                       const float* __restrict__ nomem, float* __restrict__ x) {
  int t = blockIdx.x;            // b*4096 + s
  int s = t & 4095;
  int c = threadIdx.x;
  x[(size_t)t * 256 + c] = feat[c * 4096 + s] - nomem[c] + 0.1f * pos[s * 256 + c];
}

// ---------------------------------------------------------------------------
// mem / kin assembly (f16), 3648 tokens x 64 dims per batch
// ---------------------------------------------------------------------------
__global__ void prep_mem(const float* __restrict__ mem0, const float* __restrict__ mem1,
                         const float* __restrict__ mpos, h16* __restrict__ memh,
                         h16* __restrict__ kinh) {
  int tt = blockIdx.x;           // b*3648 + t
  int b = tt / 3648, t = tt - b * 3648;
  int d = threadIdx.x;
  float mv;
  if (t < 3584) {
    int m = t >> 9, j = t & 511;
    mv = mem1[((((size_t)b * 7 + m) * 512) + j) * 64 + d];
  } else {
    int u = t - 3584, p = u >> 2, sp = u & 3;
    mv = mem0[((size_t)b * 16 + p) * 256 + sp * 64 + d];
  }
  memh[(size_t)tt * 64 + d] = (h16)mv;
  kinh[(size_t)tt * 64 + d] = (h16)(mv + mpos[(size_t)tt * 64 + d]);
}

// ---------------------------------------------------------------------------
// Axial RoPE cos/sin tables: blocks 0..4095 -> 64x64 grid, 4096.. -> 32x16 grid
// ---------------------------------------------------------------------------
__global__ void rope_tables(float* __restrict__ cq, float* __restrict__ sq,
                            float* __restrict__ csp, float* __restrict__ ssp) {
  int blk = blockIdx.x, i = threadIdx.x;     // i in [0,128)
  int j = (i < 64) ? i : (i - 64);
  float f = __powf(10000.f, -(float)j * (1.f / 64.f));
  if (blk < 4096) {
    float coord = (i < 64) ? (float)(blk & 63) : (float)(blk >> 6);
    float a = coord * f;
    cq[blk * 128 + i] = __cosf(a);
    sq[blk * 128 + i] = __sinf(a);
  } else {
    int t = blk - 4096;                      // [0,512)
    float coord = (i < 64) ? (float)(t & 31) : (float)(t >> 5);
    float a = coord * f;
    csp[t * 128 + i] = __cosf(a);
    ssp[t * 128 + i] = __sinf(a);
  }
}

// ---------------------------------------------------------------------------
// RoPE apply on f32 rows -> f16 rows. Tokens beyond nrope pass through.
// ---------------------------------------------------------------------------
__global__ void rope_apply(const float* __restrict__ in, h16* __restrict__ out,
                           const float* __restrict__ ct, const float* __restrict__ st,
                           int T, int mod, int nrope) {
  int tt = blockIdx.x;            // b*T + t
  int t = tt % T;
  int i = threadIdx.x;            // pair index [0,128)
  size_t base = (size_t)tt * 256;
  float xe = in[base + 2 * i], xo = in[base + 2 * i + 1];
  if (t < nrope) {
    int r = t % mod;
    float c = ct[r * 128 + i], s = st[r * 128 + i];
    out[base + 2 * i]     = (h16)(xe * c - xo * s);
    out[base + 2 * i + 1] = (h16)(xe * s + xo * c);
  } else {
    out[base + 2 * i]     = (h16)xe;
    out[base + 2 * i + 1] = (h16)xo;
  }
}

// ---------------------------------------------------------------------------
// LayerNorm (one wave per token, C=256) -> f16
// ---------------------------------------------------------------------------
__global__ __launch_bounds__(32) void ln_h16(const float* __restrict__ X,
                                             const float* __restrict__ gs,
                                             const float* __restrict__ gb,
                                             h16* __restrict__ out) {
  int t = blockIdx.x, lane = threadIdx.x;
  const float* xr = X + (size_t)t * 256;
  float v[8], s = 0.f;
#pragma unroll
  for (int j = 0; j < 8; ++j) { v[j] = xr[lane + 32 * j]; s += v[j]; }
#pragma unroll
  for (int d = 1; d < 32; d <<= 1) s += __shfl_xor(s, d, 32);
  float mean = s * (1.f / 256.f), q = 0.f;
#pragma unroll
  for (int j = 0; j < 8; ++j) { float d0 = v[j] - mean; q += d0 * d0; }
#pragma unroll
  for (int d = 1; d < 32; d <<= 1) q += __shfl_xor(q, d, 32);
  float r = rsqrtf(q * (1.f / 256.f) + 1e-5f);
#pragma unroll
  for (int j = 0; j < 8; ++j) {
    int c = lane + 32 * j;
    out[(size_t)t * 256 + c] = (h16)((v[j] - mean) * r * gs[c] + gb[c]);
  }
}

// Final LN + transpose to (B,C,64,64) f32 output
__global__ __launch_bounds__(32) void ln_final(const float* __restrict__ X,
                                               const float* __restrict__ gs,
                                               const float* __restrict__ gb,
                                               float* __restrict__ out) {
  int t = blockIdx.x, lane = threadIdx.x;
  int b = t >> 12, sidx = t & 4095;
  const float* xr = X + (size_t)t * 256;
  float v[8], s = 0.f;
#pragma unroll
  for (int j = 0; j < 8; ++j) { v[j] = xr[lane + 32 * j]; s += v[j]; }
#pragma unroll
  for (int d = 1; d < 32; d <<= 1) s += __shfl_xor(s, d, 32);
  float mean = s * (1.f / 256.f), q = 0.f;
#pragma unroll
  for (int j = 0; j < 8; ++j) { float d0 = v[j] - mean; q += d0 * d0; }
#pragma unroll
  for (int d = 1; d < 32; d <<= 1) q += __shfl_xor(q, d, 32);
  float r = rsqrtf(q * (1.f / 256.f) + 1e-5f);
#pragma unroll
  for (int j = 0; j < 8; ++j) {
    int c = lane + 32 * j;
    out[((size_t)(b * 256 + c)) * 4096 + sidx] = (v[j] - mean) * r * gs[c] + gb[c];
  }
}

// ---------------------------------------------------------------------------
// WMMA GEMM: Y[M,N] = X[M,K] @ W[K,N] + bias  (X f16 row-major, Wt f16 [N][K])
// One wave -> 16x64 tile. Fused relu / residual / f32 / f16 / transposed-f16.
// ---------------------------------------------------------------------------
__global__ __launch_bounds__(32) void gemm_wmma(
    const h16* __restrict__ X, const h16* __restrict__ Wt,
    const float* __restrict__ bias, const float* res, float* outF,
    h16* outH, h16* outT, int M, int N, int K, int rowsPerBatch, int flags) {
  int lane = threadIdx.x;
  int half = lane >> 4, ln = lane & 15;
  int mb = blockIdx.x * 16;
  int nb = blockIdx.y * 64;

  v8f acc[4];
#pragma unroll
  for (int t = 0; t < 4; ++t)
#pragma unroll
    for (int v = 0; v < 8; ++v) acc[t][v] = 0.f;

  const h16* xr = X + (size_t)(mb + ln) * K;
  for (int kk = 0; kk < K; kk += 32) {
    __builtin_prefetch(xr + kk + 256, 0, 3);
    __builtin_prefetch(Wt + (size_t)(nb + ln) * K + kk + 256, 0, 3);
    v16h a;
#pragma unroll
    for (int e = 0; e < 16; ++e)
      a[e] = xr[kk + e + half * 16];
#pragma unroll
    for (int t = 0; t < 4; ++t) {
      v16h bf;
      const h16* wp = Wt + (size_t)(nb + 16 * t + ln) * K + kk + half * 16;
#pragma unroll
      for (int e = 0; e < 16; ++e) bf[e] = wp[e];
      acc[t] = wmma_f16(a, bf, acc[t]);
    }
  }

#pragma unroll
  for (int t = 0; t < 4; ++t) {
    int n = nb + 16 * t + ln;
#pragma unroll
    for (int v = 0; v < 8; ++v) {
      int m = mb + half * 8 + v;
      float val = acc[t][v] + bias[n];
      if (flags & GF_RELU) val = fmaxf(val, 0.f);
      if (flags & GF_RES)  val += res[(size_t)m * N + n];
      if (flags & GF_OUTF) outF[(size_t)m * N + n] = val;
      if (flags & GF_OUTH) outH[(size_t)m * N + n] = (h16)val;
      if (flags & GF_OUTT) {
        int bb = m / rowsPerBatch;
        int mi = m - bb * rowsPerBatch;
        outT[((size_t)bb * N + n) * rowsPerBatch + mi] = (h16)val;   // [B][feat][tok]
      }
    }
  }
}

// ---------------------------------------------------------------------------
// Flash attention, async-LDS edition.
// Block = 128 threads (4 waves, 64 queries). K/V tiles for 32 keys are
// double-buffered in LDS via global_load_async_to_lds_b128 (ASYNCcnt), shared
// by all 4 waves; each wave runs 16 score WMMAs + online softmax + 16 P.V WMMAs
// per tile. Q,K row-major f16 [B][S][256]; V transposed f16 [B][256][Skv].
// ---------------------------------------------------------------------------
__global__ __launch_bounds__(128) void attn_wmma(
    const h16* __restrict__ Q, const h16* __restrict__ Kx,
    const h16* __restrict__ Vt, h16* __restrict__ O,
    int Sq, int Skv, float scale) {
  __shared__ __align__(16) h16 kbuf[2][32 * 256];   // [key][feat]
  __shared__ __align__(16) h16 vbuf[2][256 * 32];   // [feat][key]
  __shared__ __align__(16) h16 Pb[4][16][32];

  int tid = threadIdx.x;
  int w = tid >> 5;
  int lane = tid & 31;
  int half = lane >> 4, ln = lane & 15;
  int b = blockIdx.y;
  int qb = blockIdx.x * 64 + w * 16;
  const h16* Qb = Q + (size_t)b * Sq * 256;
  const h16* Kb = Kx + (size_t)b * Skv * 256;
  const h16* Vb = Vt + (size_t)b * 256 * Skv;
  h16* Ob = O + (size_t)b * Sq * 256;

  // stage one 32-key K tile (16KB) + V tile (16KB): 16 async ops per thread
  auto stage = [&](int buf, int kb0) {
#pragma unroll
    for (int i = 0; i < 8; ++i) {
      int chunk = i * 128 + tid;               // 1024 chunks of 16B
      int row = chunk >> 5, c8 = (chunk & 31) << 3;
      async_cp16(Kb + (size_t)(kb0 + row) * 256 + c8,
                 lds_lo32(&kbuf[buf][(row << 8) + c8]));
    }
#pragma unroll
    for (int i = 0; i < 8; ++i) {
      int chunk = i * 128 + tid;               // 1024 chunks of 16B
      int feat = chunk >> 2, c8 = (chunk & 3) << 3;
      async_cp16(Vb + (size_t)feat * Skv + kb0 + c8,
                 lds_lo32(&vbuf[buf][(feat << 5) + c8]));
    }
  };

  // Q fragments (scale folded in), 8 frags cover head dim 256
  v16h qa[8];
#pragma unroll
  for (int f = 0; f < 8; ++f) {
    const h16* rp = Qb + (size_t)(qb + ln) * 256 + f * 32 + half * 16;
#pragma unroll
    for (int e = 0; e < 16; ++e)
      qa[f][e] = (h16)((float)rp[e] * scale);
  }

  v8f acc[16];
#pragma unroll
  for (int t = 0; t < 16; ++t)
#pragma unroll
    for (int v = 0; v < 8; ++v) acc[t][v] = 0.f;
  float mrun[8], lrun[8];
#pragma unroll
  for (int v = 0; v < 8; ++v) { mrun[v] = -1e30f; lrun[v] = 0.f; }

  int nsteps = Skv / 32;
  stage(0, 0);

  for (int s = 0; s < nsteps; ++s) {
    int cur = s & 1;
    if (s + 1 < nsteps) {
      stage(cur ^ 1, (s + 1) * 32);
      asm volatile("s_wait_asynccnt 0x10" ::: "memory");  // stage(s) retired
    } else {
      asm volatile("s_wait_asynccnt 0x0" ::: "memory");
    }
    __syncthreads();                                      // tile valid block-wide

    const h16* ks = kbuf[cur];
    const h16* vs = vbuf[cur];

    v8f s0, s1;
#pragma unroll
    for (int v = 0; v < 8; ++v) { s0[v] = 0.f; s1[v] = 0.f; }
#pragma unroll
    for (int f = 0; f < 8; ++f) {
      v16h b0, b1;
      const h16* kp0 = ks + (size_t)ln * 256 + f * 32 + half * 16;
      const h16* kp1 = ks + (size_t)(16 + ln) * 256 + f * 32 + half * 16;
#pragma unroll
      for (int e = 0; e < 16; ++e) { b0[e] = kp0[e]; b1[e] = kp1[e]; }
      s0 = wmma_f16(qa[f], b0, s0);
      s1 = wmma_f16(qa[f], b1, s1);
    }

    // online softmax (rows 8*half+v live across this 16-lane group)
    float alpha[8];
#pragma unroll
    for (int v = 0; v < 8; ++v) {
      float mx = fmaxf(s0[v], s1[v]);
#pragma unroll
      for (int d = 1; d < 16; d <<= 1) mx = fmaxf(mx, __shfl_xor(mx, d, 32));
      float mnew = fmaxf(mrun[v], mx);
      float p0 = __expf(s0[v] - mnew);
      float p1 = __expf(s1[v] - mnew);
      float a = __expf(mrun[v] - mnew);
      float rs = p0 + p1;
#pragma unroll
      for (int d = 1; d < 16; d <<= 1) rs += __shfl_xor(rs, d, 32);
      lrun[v] = lrun[v] * a + rs;
      mrun[v] = mnew;
      alpha[v] = a;
      Pb[w][half * 8 + v][ln]      = (h16)p0;
      Pb[w][half * 8 + v][16 + ln] = (h16)p1;
    }
#pragma unroll
    for (int t = 0; t < 16; ++t)
#pragma unroll
      for (int v = 0; v < 8; ++v) acc[t][v] *= alpha[v];

    asm volatile("s_wait_dscnt 0" ::: "memory");   // P stores visible in-wave

    v16h pa;
#pragma unroll
    for (int e = 0; e < 16; ++e)
      pa[e] = Pb[w][ln][e + half * 16];

#pragma unroll
    for (int t = 0; t < 16; ++t) {
      v16h vb;
      const h16* vp = vs + (size_t)(16 * t + ln) * 32 + half * 16;
#pragma unroll
      for (int e = 0; e < 16; ++e) vb[e] = vp[e];
      acc[t] = wmma_f16(pa, vb, acc[t]);
    }

    asm volatile("s_wait_dscnt 0" ::: "memory");   // LDS reads done
    __syncthreads();                               // safe to overwrite buffers
  }

  float inv[8];
#pragma unroll
  for (int v = 0; v < 8; ++v) inv[v] = 1.0f / lrun[v];
#pragma unroll
  for (int t = 0; t < 16; ++t)
#pragma unroll
    for (int v = 0; v < 8; ++v)
      Ob[(size_t)(qb + half * 8 + v) * 256 + 16 * t + ln] = (h16)(acc[t][v] * inv[v]);
}

// ---------------------------------------------------------------------------
extern "C" void kernel_launch(void* const* d_in, const int* in_sizes, int n_in,
                              void* d_out, int out_size, void* d_ws, size_t ws_size,
                              hipStream_t stream) {
  (void)in_sizes; (void)n_in; (void)out_size; (void)ws_size;
  const float* feat  = (const float*)d_in[0];
  const float* qpos  = (const float*)d_in[1];
  const float* mem0  = (const float*)d_in[2];
  const float* mem1  = (const float*)d_in[3];
  const float* mpos  = (const float*)d_in[4];
  const float* nomem = (const float*)d_in[5];
  const float* w_sa_q = (const float*)d_in[6];
  const float* w_sa_k = (const float*)d_in[7];
  const float* w_sa_v = (const float*)d_in[8];
  const float* w_sa_o = (const float*)d_in[9];
  const float* w_ca_q = (const float*)d_in[10];
  const float* w_ca_k = (const float*)d_in[11];
  const float* w_ca_v = (const float*)d_in[12];
  const float* w_ca_o = (const float*)d_in[13];
  const float* w_w1   = (const float*)d_in[14];
  const float* w_w2   = (const float*)d_in[15];
  const float* b_sa_q = (const float*)d_in[16];
  const float* b_sa_k = (const float*)d_in[17];
  const float* b_sa_v = (const float*)d_in[18];
  const float* b_sa_o = (const float*)d_in[19];
  const float* b_ca_q = (const float*)d_in[20];
  const float* b_ca_k = (const float*)d_in[21];
  const float* b_ca_v = (const float*)d_in[22];
  const float* b_ca_o = (const float*)d_in[23];
  const float* b_b1   = (const float*)d_in[24];
  const float* b_b2   = (const float*)d_in[25];
  const float* n1_s = (const float*)d_in[26];
  const float* n2_s = (const float*)d_in[27];
  const float* n3_s = (const float*)d_in[28];
  const float* n1_b = (const float*)d_in[29];
  const float* n2_b = (const float*)d_in[30];
  const float* n3_b = (const float*)d_in[31];
  const float* nf_s = (const float*)d_in[32];
  const float* nf_b = (const float*)d_in[33];
  float* out = (float*)d_out;

  const int B = 2, Sq = 4096, C = 256, Dff = 2048, Tkv = 3648, L = 4;
  const float scale = 0.0625f;

  char* ws = (char*)d_ws;
  size_t off = 0;
  auto take = [&](size_t bytes) -> char* {
    char* p = ws + off;
    off = (off + bytes + 255) & ~(size_t)255;
    return p;
  };
  float* x     = (float*)take((size_t)B * Sq * C * 4);        // residual f32
  h16*   t2h   = (h16*)  take((size_t)B * Sq * C * 2);
  float* projF = (float*)take((size_t)B * Sq * C * 4);        // >= B*Tkv*C*4
  h16*   qh    = (h16*)  take((size_t)B * Sq * C * 2);
  h16*   kh    = (h16*)  take((size_t)B * Sq * C * 2);        // >= B*Tkv*C
  h16*   vth   = (h16*)  take((size_t)B * Sq * C * 2);        // transposed V
  h16*   atth  = (h16*)  take((size_t)B * Sq * C * 2);
  h16*   memh  = (h16*)  take((size_t)B * Tkv * 64 * 2);
  h16*   kinh  = (h16*)  take((size_t)B * Tkv * 64 * 2);
  h16*   hidh  = (h16*)  take((size_t)2048 * Dff * 2);        // MLP chunk (2048 rows)
  float* cosq  = (float*)take((size_t)4096 * 128 * 4);
  float* sinq  = (float*)take((size_t)4096 * 128 * 4);
  float* cossp = (float*)take((size_t)512 * 128 * 4);
  float* sinsp = (float*)take((size_t)512 * 128 * 4);
  h16*   wts   = (h16*)  take((size_t)L * 1474560 * 2);

  // per-layer f16 weight slab offsets (elements)
  const size_t WSAQ = 0, WSAK = 65536, WSAV = 131072, WSAO = 196608, WCAQ = 262144,
               WCAK = 327680, WCAV = 344064, WCAO = 360448, WW1 = 425984, WW2 = 950272,
               WPL = 1474560;

  // --- pre-pass -------------------------------------------------------------
  rope_tables<<<dim3(4096 + 512), dim3(128), 0, stream>>>(cosq, sinq, cossp, sinsp);
  prep_x<<<dim3(B * Sq), dim3(256), 0, stream>>>(feat, qpos, nomem, x);
  prep_mem<<<dim3(B * Tkv), dim3(64), 0, stream>>>(mem0, mem1, mpos, memh, kinh);

  for (int i = 0; i < L; ++i) {
    h16* wl = wts + (size_t)i * WPL;
    wt_convert<<<dim3(256), dim3(256), 0, stream>>>(w_sa_q + (size_t)i * 65536, wl + WSAQ, 256, 256);
    wt_convert<<<dim3(256), dim3(256), 0, stream>>>(w_sa_k + (size_t)i * 65536, wl + WSAK, 256, 256);
    wt_convert<<<dim3(256), dim3(256), 0, stream>>>(w_sa_v + (size_t)i * 65536, wl + WSAV, 256, 256);
    wt_convert<<<dim3(256), dim3(256), 0, stream>>>(w_sa_o + (size_t)i * 65536, wl + WSAO, 256, 256);
    wt_convert<<<dim3(256), dim3(256), 0, stream>>>(w_ca_q + (size_t)i * 65536, wl + WCAQ, 256, 256);
    wt_convert<<<dim3(64),  dim3(256), 0, stream>>>(w_ca_k + (size_t)i * 16384, wl + WCAK, 64, 256);
    wt_convert<<<dim3(64),  dim3(256), 0, stream>>>(w_ca_v + (size_t)i * 16384, wl + WCAV, 64, 256);
    wt_convert<<<dim3(256), dim3(256), 0, stream>>>(w_ca_o + (size_t)i * 65536, wl + WCAO, 256, 256);
    wt_convert<<<dim3(2048), dim3(256), 0, stream>>>(w_w1 + (size_t)i * 524288, wl + WW1, 256, 2048);
    wt_convert<<<dim3(2048), dim3(256), 0, stream>>>(w_w2 + (size_t)i * 524288, wl + WW2, 2048, 256);
  }

  const int Mq = B * Sq;     // 8192
  const int Mk = B * Tkv;    // 7296
  dim3 gq(Mq / 16, C / 64), gk(Mk / 16, C / 64), blk(32);
  dim3 ga(Sq / 64, B), blka(128);

  // --- layers ---------------------------------------------------------------
  for (int i = 0; i < L; ++i) {
    h16* wl = wts + (size_t)i * WPL;

    // ---- self attention ----
    ln_h16<<<dim3(Mq), blk, 0, stream>>>(x, n1_s + i * 256, n1_b + i * 256, t2h);
    gemm_wmma<<<gq, blk, 0, stream>>>(t2h, wl + WSAQ, b_sa_q + i * 256, nullptr, projF,
                                      nullptr, nullptr, Mq, C, C, Sq, GF_OUTF);
    rope_apply<<<dim3(Mq), dim3(128), 0, stream>>>(projF, qh, cosq, sinq, Sq, 4096, 4096);
    gemm_wmma<<<gq, blk, 0, stream>>>(t2h, wl + WSAK, b_sa_k + i * 256, nullptr, projF,
                                      nullptr, nullptr, Mq, C, C, Sq, GF_OUTF);
    rope_apply<<<dim3(Mq), dim3(128), 0, stream>>>(projF, kh, cosq, sinq, Sq, 4096, 4096);
    gemm_wmma<<<gq, blk, 0, stream>>>(t2h, wl + WSAV, b_sa_v + i * 256, nullptr, nullptr,
                                      nullptr, vth, Mq, C, C, Sq, GF_OUTT);
    attn_wmma<<<ga, blka, 0, stream>>>(qh, kh, vth, atth, Sq, Sq, scale);
    gemm_wmma<<<gq, blk, 0, stream>>>(atth, wl + WSAO, b_sa_o + i * 256, x, x,
                                      nullptr, nullptr, Mq, C, C, Sq, GF_RES | GF_OUTF);

    // ---- cross attention ----
    ln_h16<<<dim3(Mq), blk, 0, stream>>>(x, n2_s + i * 256, n2_b + i * 256, t2h);
    gemm_wmma<<<gq, blk, 0, stream>>>(t2h, wl + WCAQ, b_ca_q + i * 256, nullptr, projF,
                                      nullptr, nullptr, Mq, C, C, Sq, GF_OUTF);
    rope_apply<<<dim3(Mq), dim3(128), 0, stream>>>(projF, qh, cosq, sinq, Sq, 4096, 4096);
    gemm_wmma<<<gk, blk, 0, stream>>>(kinh, wl + WCAK, b_ca_k + i * 256, nullptr, projF,
                                      nullptr, nullptr, Mk, C, 64, Tkv, GF_OUTF);
    rope_apply<<<dim3(Mk), dim3(128), 0, stream>>>(projF, kh, cossp, sinsp, Tkv, 512, 3584);
    gemm_wmma<<<gk, blk, 0, stream>>>(memh, wl + WCAV, b_ca_v + i * 256, nullptr, nullptr,
                                      nullptr, vth, Mk, C, 64, Tkv, GF_OUTT);
    attn_wmma<<<ga, blka, 0, stream>>>(qh, kh, vth, atth, Sq, Tkv, scale);
    gemm_wmma<<<gq, blk, 0, stream>>>(atth, wl + WCAO, b_ca_o + i * 256, x, x,
                                      nullptr, nullptr, Mq, C, C, Sq, GF_RES | GF_OUTF);

    // ---- MLP (chunked over rows to bound workspace) ----
    ln_h16<<<dim3(Mq), blk, 0, stream>>>(x, n3_s + i * 256, n3_b + i * 256, t2h);
    for (int m0 = 0; m0 < Mq; m0 += 2048) {
      gemm_wmma<<<dim3(2048 / 16, Dff / 64), blk, 0, stream>>>(
          t2h + (size_t)m0 * C, wl + WW1, b_b1 + i * Dff, nullptr, nullptr,
          hidh, nullptr, 2048, Dff, C, Sq, GF_RELU | GF_OUTH);
      gemm_wmma<<<dim3(2048 / 16, C / 64), blk, 0, stream>>>(
          hidh, wl + WW2, b_b2 + i * 256, x + (size_t)m0 * C, x + (size_t)m0 * C,
          nullptr, nullptr, 2048, C, Dff, Sq, GF_RES | GF_OUTF);
    }
  }

  // --- final LN + transpose to (B,C,64,64) ----------------------------------
  ln_final<<<dim3(Mq), blk, 0, stream>>>(x, nf_s, nf_b, out);
}
